// Decoder_5136780886515
// MI455X (gfx1250) — matrix operand order repaired and checked by reading, not compiled, not run on hardware
//
#include <hip/hip_runtime.h>
#include <stdint.h>

#define VOCAB 32000
#define EM    256
#define UNITS 512
#define BB    64
#define TT    16
#define G4U   (4*UNITS)    // 2048
#define KOUT  (TT*UNITS)   // 8192

typedef __bf16 bf16_t;
typedef bf16_t v16bf __attribute__((ext_vector_type(16)));
typedef float  v8f   __attribute__((ext_vector_type(8)));
typedef uint32_t u32x8 __attribute__((ext_vector_type(8)));
typedef uint32_t u32x4 __attribute__((ext_vector_type(4)));
typedef int      i32x8 __attribute__((ext_vector_type(8)));
typedef int      i32x4 __attribute__((ext_vector_type(4)));

// Pack two f32 -> one dword of two bf16 (s0 -> low half). Single VALU op.
__device__ __forceinline__ uint32_t pk_bf16(float lo, float hi) {
#if __has_builtin(__builtin_amdgcn_cvt_pk_bf16_f32)
    auto p = __builtin_amdgcn_cvt_pk_bf16_f32(lo, hi);
    uint32_t r; __builtin_memcpy(&r, &p, 4);
    return r;
#else
    uint32_t r;
    asm("v_cvt_pk_bf16_f32 %0, %1, %2" : "=v"(r) : "v"(lo), "v"(hi));
    return r;
#endif
}

__device__ __forceinline__ float sigmoidf(float x) {
    return 1.0f / (1.0f + __expf(-x));
}

// B tile in LDS: 32 rows x 128 f32, TDM pads 2 dwords per 128 -> stride 130
// (16*130 mod 64 = 32: half-wave column reads land on disjoint banks)
#define BSTR 130
// A tile in LDS: 64 rows x 16 dwords of packed bf16 K-pairs, stride 20
// (16B-aligned rows for b128 LDS ops; 20*l mod 64 distinct for 16 consecutive rows)
#define ASTR 20

// C[m0:m0+64, n0:n0+128] = A[m0:,:K] * B[:K, n0:] (+ addend[m,n]) (+ bias[n])
// One block owns ALL 64 M-rows: 4 A-fragments, 1 shared B-fragment, 4 WMMAs/tile.
// GATHER: A row m comes from table[tokens[b*T+t]*EM], m = t*64+b.
template <bool GATHER>
__global__ __launch_bounds__(256) void gemm_wmma_bf16(
    const float* __restrict__ A, int lda,
    const float* __restrict__ Bm, int ldb,
    float* __restrict__ C, int ldc, int K,
    const float* __restrict__ addend,   // same row/col indexing as C, or null
    const float* __restrict__ bias,     // per-column, or null
    const int*   __restrict__ tokens,   // GATHER only
    const float* __restrict__ table)    // GATHER only
{
    __shared__ uint32_t sA[64][ASTR];       // packed bf16 K-pairs
    __shared__ float    sB[2][32 * BSTR];   // double-buffered f32 tiles (TDM dest)
    __shared__ int      sTok[64];

    const int tid  = threadIdx.x;
    const int lane = tid & 31;
    const int wave = tid >> 5;                 // 0..7, each owns one 16-col N tile
    const int m0   = blockIdx.y * 64;
    const int n0   = blockIdx.x * 128;

    if (GATHER) {
        if (tid < 64) {
            int m = m0 + tid;
            int b = m & (BB - 1);
            int t = m >> 6;
            sTok[tid] = tokens[b * TT + t];
        }
        __syncthreads();
    }

    // Per-thread A staging assignment (hoisted): row r = tid>>2, 8 floats at cc
    const int arow_r  = tid >> 2;
    const int arow_cc = (tid & 3) * 8;
    const float* arow = GATHER ? (table + (size_t)sTok[arow_r] * EM)
                               : (A + (size_t)(m0 + arow_r) * lda);

    // ---- TDM descriptor group1 (uniform over all iterations), ISA 8.4:
    //  d0: data_size=2 (4B) | pad_enable | pad_interval=6 (128 dw) | pad_amount=1 (2 dw)
    //  d1[31:16]=tensor_dim0 lo (128)   d2[31:16]=tensor_dim1 lo (32)
    //  d3[31:16]=tile_dim0 (128)        d4[15:0]=tile_dim1 (32)
    //  d5=tensor_dim0_stride lo (ldb elements)
    i32x8 g1v;
    g1v[0] = (2 << 16) | (1 << 20) | (6 << 22) | (1 << 25);
    g1v[1] = (128 << 16);
    g1v[2] = (32 << 16);
    g1v[3] = (128 << 16);
    g1v[4] = 32;
    g1v[5] = ldb;
    g1v[6] = 0;
    g1v[7] = 0;
    const i32x4 gz4 = {0, 0, 0, 0};
    const i32x8 gz8 = {0, 0, 0, 0, 0, 0, 0, 0};

    auto tdm_issue = [&](int kk, int bufIdx) {
        uint64_t ga  = (uint64_t)(uintptr_t)(Bm + (size_t)kk * ldb + n0);
        uint32_t lds = (uint32_t)(uintptr_t)&sB[bufIdx][0];
        u32x4 g0;
        g0[0] = 1u;                                   // count=1, user descriptor
        g0[1] = lds;                                  // lds_addr
        g0[2] = (uint32_t)ga;                         // global_addr[31:0]
        g0[3] = (uint32_t)(ga >> 32) | (2u << 30);    // global_addr[56:32] | type=2
        __builtin_amdgcn_tensor_load_to_lds(g0, g1v, gz4, gz4, gz8, 0);
    };

    // prologue: kick tile 0
    if (wave == 0) tdm_issue(0, 0);

    v8f acc[4] = {{}, {}, {}, {}};
    int buf = 0;

    for (int k0 = 0; k0 < K; k0 += 32) {
        // ---- stage A tile (64x32 f32 -> packed bf16 K-pair dwords), all 256 threads
        {
            float4 a0 = *(const float4*)(arow + k0 + arow_cc);
            float4 a1 = *(const float4*)(arow + k0 + arow_cc + 4);
            uint4 pkd;
            pkd.x = pk_bf16(a0.x, a0.y);
            pkd.y = pk_bf16(a0.z, a0.w);
            pkd.z = pk_bf16(a1.x, a1.y);
            pkd.w = pk_bf16(a1.z, a1.w);
            *(uint4*)&sA[arow_r][arow_cc >> 1] = pkd;   // ds_store_b128
        }
        // ---- issue next B tile, then wait for the current one (per-wave in-order TENSORcnt)
        if (wave == 0) {
            if (k0 + 32 < K) {
                tdm_issue(k0 + 32, buf ^ 1);
                __builtin_amdgcn_s_wait_tensorcnt((short)1);
            } else {
                __builtin_amdgcn_s_wait_tensorcnt((short)0);
            }
        }
        __syncthreads();

        // ---- fragments (contiguous K order k = 16*(lane>>4) + s, same for A and B)
        const int l   = lane & 15;
        const int sel = lane >> 4;
        // B: lanes 0-15 hold K=0..15 of column N=lane; lanes 16-31 hold K=16..31
        const int col  = wave * 16 + l;
        const int koff = sel * 16;
        const float* Bb = &sB[buf][0];
        u32x8 bu;
        #pragma unroll
        for (int i = 0; i < 8; ++i)
            bu[i] = pk_bf16(Bb[(koff + 2 * i) * BSTR + col],
                            Bb[(koff + 2 * i + 1) * BSTR + col]);
        v16bf bf = __builtin_bit_cast(v16bf, bu);

        // A: lanes<16 hold row M=row, K 0..15; lanes>=16 K 16..31
        #pragma unroll
        for (int mi = 0; mi < 4; ++mi) {
            const int row = mi * 16 + l;
            uint4 a0 = *(const uint4*)&sA[row][8 * sel];        // ds_load_b128
            uint4 a1 = *(const uint4*)&sA[row][8 * sel + 4];    // ds_load_b128
            u32x8 au;
            au[0] = a0.x; au[1] = a0.y; au[2] = a0.z; au[3] = a0.w;
            au[4] = a1.x; au[5] = a1.y; au[6] = a1.z; au[7] = a1.w;
            v16bf af = __builtin_bit_cast(v16bf, au);
            acc[mi] = __builtin_amdgcn_wmma_f32_16x16x32_bf16(false, af, false, bf,
                                                              (short)0, acc[mi], false, false);
        }
        __syncthreads();   // protect sA and retire reads of sB[buf] before it is re-filled
        buf ^= 1;
    }

    // ---- store C: VGPR r <-> M = mi*16 + r + 8*(lane>=16), N = lane&15
    const int rowh = (lane >> 4) * 8;
    const int colo = n0 + wave * 16 + (lane & 15);
    const float badd = bias ? bias[colo] : 0.0f;
    #pragma unroll
    for (int mi = 0; mi < 4; ++mi) {
        #pragma unroll
        for (int r = 0; r < 8; ++r) {
            int mr = m0 + mi * 16 + rowh + r;
            float v = acc[mi][r] + badd;
            if (addend) v += addend[(size_t)mr * ldc + colo];
            C[(size_t)mr * ldc + colo] = v;
        }
    }
}

__global__ void init_hc(const float* __restrict__ eh, const float* __restrict__ ec,
                        float* __restrict__ h, float* __restrict__ c) {
    int i = blockIdx.x * blockDim.x + threadIdx.x;
    if (i < BB * UNITS) { h[i] = eh[i]; c[i] = ec[i]; }
}

// z:[64][2048] (i,f,g,o in 512-col bands, all sigmoid per reference); updates h,c; writes flat
__global__ void lstm_pointwise(const float* __restrict__ z, float* __restrict__ h,
                               float* __restrict__ c, float* __restrict__ flat, int t) {
    int i = blockIdx.x * blockDim.x + threadIdx.x;
    if (i >= BB * UNITS) return;
    int b = i / UNITS;
    int u = i - b * UNITS;
    const float* zr = z + (size_t)b * G4U;
    float ig = sigmoidf(zr[u]);
    float fg = sigmoidf(zr[UNITS + u]);
    float gg = sigmoidf(zr[2 * UNITS + u]);
    float og = sigmoidf(zr[3 * UNITS + u]);
    float cn = fg * c[i] + ig * gg;
    float hn = og * sigmoidf(cn);
    c[i] = cn;
    h[i] = hn;
    flat[(size_t)b * KOUT + (size_t)t * UNITS + u] = hn;
}

__global__ __launch_bounds__(256) void softmax_rows(const float* __restrict__ logits,
                                                    float* __restrict__ out) {
    __shared__ float red[256];
    const int row = blockIdx.x;
    const float* x = logits + (size_t)row * VOCAB;
    float m = -__builtin_inff();
    for (int j = threadIdx.x; j < VOCAB; j += 256) m = fmaxf(m, x[j]);
    red[threadIdx.x] = m; __syncthreads();
    for (int s = 128; s > 0; s >>= 1) {
        if (threadIdx.x < s) red[threadIdx.x] = fmaxf(red[threadIdx.x], red[threadIdx.x + s]);
        __syncthreads();
    }
    m = red[0]; __syncthreads();
    float sum = 0.0f;
    for (int j = threadIdx.x; j < VOCAB; j += 256) sum += __expf(x[j] - m);
    red[threadIdx.x] = sum; __syncthreads();
    for (int s = 128; s > 0; s >>= 1) {
        if (threadIdx.x < s) red[threadIdx.x] += red[threadIdx.x + s];
        __syncthreads();
    }
    const float inv = 1.0f / red[0];
    for (int j = threadIdx.x; j < VOCAB; j += 256)
        out[(size_t)row * VOCAB + j] = __expf(x[j] - m) * inv;
}

extern "C" void kernel_launch(void* const* d_in, const int* in_sizes, int n_in,
                              void* d_out, int out_size, void* d_ws, size_t ws_size,
                              hipStream_t stream) {
    const int*   inputs = (const int*)d_in[0];    // [64][16]
    const float* enc_h  = (const float*)d_in[1];  // [64][512]
    const float* enc_c  = (const float*)d_in[2];  // [64][512]
    const float* emb    = (const float*)d_in[3];  // [32000][256]
    const float* Wx     = (const float*)d_in[4];  // [256][2048]
    const float* Wh     = (const float*)d_in[5];  // [512][2048]
    const float* bvec   = (const float*)d_in[6];  // [2048]
    const float* Wout   = (const float*)d_in[7];  // [8192][32000]
    const float* bout   = (const float*)d_in[8];  // [32000]
    float* out = (float*)d_out;                   // [64][32000] softmax

    float* ws     = (float*)d_ws;
    float* xz     = ws;                               // [16*64][2048]
    float* zbuf   = xz   + (size_t)TT * BB * G4U;     // [64][2048]
    float* hbuf   = zbuf + (size_t)BB * G4U;          // [64][512]
    float* cbuf   = hbuf + (size_t)BB * UNITS;        // [64][512]
    float* flat   = cbuf + (size_t)BB * UNITS;        // [64][8192]
    float* logits = flat + (size_t)BB * KOUT;         // [64][32000]

    // 1) xz[t*64+b][:] = emb[inputs[b][t]] @ Wx + b   (each y-block = one timestep)
    gemm_wmma_bf16<true><<<dim3(G4U / 128, (BB * TT) / 64), 256, 0, stream>>>(
        nullptr, 0, Wx, G4U, xz, G4U, EM, nullptr, bvec, inputs, emb);

    init_hc<<<(BB * UNITS + 255) / 256, 256, 0, stream>>>(enc_h, enc_c, hbuf, cbuf);

    // 2) sequential LSTM scan: stream order serializes the recurrence
    for (int t = 0; t < TT; ++t) {
        const float* xz_t = xz + (size_t)t * BB * G4U;
        gemm_wmma_bf16<false><<<dim3(G4U / 128, 1), 256, 0, stream>>>(
            hbuf, UNITS, Wh, G4U, zbuf, G4U, UNITS, xz_t, nullptr, nullptr, nullptr);
        lstm_pointwise<<<(BB * UNITS + 255) / 256, 256, 0, stream>>>(zbuf, hbuf, cbuf, flat, t);
    }

    // 3) logits = flat @ Wout + bout : Wout streamed exactly once (1.05 GB -> ~45 us floor)
    gemm_wmma_bf16<false><<<dim3(VOCAB / 128, 1), 256, 0, stream>>>(
        flat, KOUT, Wout, VOCAB, logits, VOCAB, KOUT, nullptr, bout, nullptr, nullptr);

    // 4) row softmax
    softmax_rows<<<BB, 256, 0, stream>>>(logits, out);
}
